// TransformerMultiView_10307921511051
// MI455X (gfx1250) — matrix operand order repaired and checked by reading, not compiled, not run on hardware
//
#include <hip/hip_runtime.h>
#include <hip/hip_bf16.h>

// ---------------------------------------------------------------- constants
#define NTOK   1536
#define DMODEL 1024
#define NH     16
#define DH     64
#define NLAYER 4
#define VOC    8192
#define NCOND  512
#define FFIN   2730          // INNER
#define FF2IN  5460          // 2*INNER
#define QKSCALE 8.0f
#define NK_SA   1537
#define NK_CA   513
#define NKP_SA  1568         // pad to /32
#define NKP_CA  544

typedef __bf16 bf16;
typedef __attribute__((ext_vector_type(16))) __bf16 v16bf;
typedef __attribute__((ext_vector_type(8)))  __bf16 v8bf;
typedef __attribute__((ext_vector_type(2)))  __bf16 v2bf;
typedef __attribute__((ext_vector_type(8)))  float  v8f;

__device__ __forceinline__ v16bf cat8(v8bf lo, v8bf hi) {
  return __builtin_shufflevector(lo, hi, 0,1,2,3,4,5,6,7,8,9,10,11,12,13,14,15);
}
__device__ __forceinline__ v2bf pk2(float x, float y) {   // -> v_cvt_pk_bf16_f32
  v2bf r; r[0] = (__bf16)x; r[1] = (__bf16)y; return r;
}

// ---------------------------------------------------------------- embedding
__global__ __launch_bounds__(256)
void embed_kernel(const int* __restrict__ ids, const float* __restrict__ emb,
                  const float* __restrict__ pos, float* __restrict__ out) {
  const int i = blockIdx.x;
  const long row = (long)ids[i] * DMODEL;
  for (int d = threadIdx.x; d < DMODEL; d += 256)
    out[(long)i * DMODEL + d] = emb[row + d] + pos[(long)i * DMODEL + d];
}

// ---------------------------------------------------------------- layernorm
__global__ __launch_bounds__(256)
void ln_kernel(const float* __restrict__ X, const float* __restrict__ G,
               float* __restrict__ Y, int Dlen) {
  const long base = (long)blockIdx.x * Dlen;
  float s = 0.f, s2 = 0.f;
  for (int i = threadIdx.x; i < Dlen; i += 256) {
    float v = X[base + i]; s += v; s2 += v * v;
  }
  __shared__ float red[2][8];
  for (int off = 16; off; off >>= 1) {
    s  += __shfl_down(s,  off, 32);
    s2 += __shfl_down(s2, off, 32);
  }
  const int wave = threadIdx.x >> 5, lane = threadIdx.x & 31;
  if (lane == 0) { red[0][wave] = s; red[1][wave] = s2; }
  __syncthreads();
  if (wave == 0) {
    s  = (lane < 8) ? red[0][lane] : 0.f;
    s2 = (lane < 8) ? red[1][lane] : 0.f;
    for (int off = 4; off; off >>= 1) {
      s  += __shfl_down(s,  off, 32);
      s2 += __shfl_down(s2, off, 32);
    }
    if (lane == 0) { red[0][0] = s; red[1][0] = s2; }
  }
  __syncthreads();
  const float mean = red[0][0] / (float)Dlen;
  const float var  = red[1][0] / (float)Dlen - mean * mean;
  const float rstd = rsqrtf(var + 1e-5f);
  for (int i = threadIdx.x; i < Dlen; i += 256)
    Y[base + i] = (X[base + i] - mean) * rstd * G[i];
}

// ---------------------------------------------------------------- WMMA GEMM
// C[M,N] = A[M,K] @ B[K,N] (+Cin), f32 in/out, bf16 WMMA compute.
// Double-buffered LDS, register-staged global loads, branch-free interior.
#define GBM 64
#define GBN 128
#define GBK 32
__global__ __launch_bounds__(256)
void gemm_bf16_kernel(const float* __restrict__ A, const float* __restrict__ B,
                      float* __restrict__ C, const float* __restrict__ Cin,
                      int M, int N, int K, int lda, int ldb, int ldc) {
  __shared__ __bf16 As[2][GBM][GBK] __attribute__((aligned(64)));
  __shared__ __bf16 Bs[2][GBN][GBK] __attribute__((aligned(64)));   // [n][k]
  const int tid  = threadIdx.x;
  const int lane = tid & 31;
  const int wave = tid >> 5;
  const int m0 = blockIdx.y * GBM;
  const int n0 = blockIdx.x * GBN;
  const int wm = (wave >> 2) * 32;     // 0 / 32
  const int wn = (wave & 3) * 32;      // 0..96
  const int g   = lane >> 4;
  const int l15 = lane & 15;
  v8f acc[2][2] = {};

  // per-thread staging coordinates
  const int arr = tid >> 4;            // A row base 0..15 (+i*16)
  const int arc = (tid & 15) * 2;      // A k offset (float2)
  const int brk = (tid >> 5) * 2;      // B k base: even pair 0..14 (+i*16)
  const int brn = (tid & 31) * 4;      // B n offset (float4)

  auto computeTile = [&](int b) {
    v16bf afr[2], bfr[2];
    #pragma unroll
    for (int i = 0; i < 2; ++i) {      // A frag: K in {g*8..+7, 16+g*8..+7}
      const int mr = wm + i * 16 + l15;
      v8bf lo = *(const v8bf*)&As[b][mr][g * 8];
      v8bf hi = *(const v8bf*)&As[b][mr][16 + g * 8];
      afr[i] = cat8(lo, hi);
    }
    #pragma unroll
    for (int j = 0; j < 2; ++j)        // B frag: K in {g*8..+7, 16+g*8..+7}
      bfr[j] = cat8(*(const v8bf*)&Bs[b][wn + j * 16 + l15][g * 8],
                    *(const v8bf*)&Bs[b][wn + j * 16 + l15][16 + g * 8]);
    #pragma unroll
    for (int i = 0; i < 2; ++i)
      #pragma unroll
      for (int j = 0; j < 2; ++j)
        acc[i][j] = __builtin_amdgcn_wmma_f32_16x16x32_bf16(
            false, afr[i], false, bfr[j], (short)0, acc[i][j], false, false);
  };

  if (m0 + GBM <= M && n0 + GBN <= N) {
    // ---------------- fast interior path ----------------
    float2 aReg[4];
    float4 bReg[4];                    // pairs: (k, k+1) x 2 k-groups
    auto loadRegs = [&](int kb) {
      #pragma unroll
      for (int i = 0; i < 4; ++i)
        aReg[i] = *(const float2*)&A[(long)(m0 + arr + i * 16) * lda + kb + arc];
      #pragma unroll
      for (int i = 0; i < 2; ++i) {
        bReg[2 * i + 0] = *(const float4*)&B[(long)(kb + brk + i * 16 + 0) * ldb + n0 + brn];
        bReg[2 * i + 1] = *(const float4*)&B[(long)(kb + brk + i * 16 + 1) * ldb + n0 + brn];
      }
    };
    auto storeRegs = [&](int b) {
      #pragma unroll
      for (int i = 0; i < 4; ++i)      // one pk-cvt + one b32 DS store each
        *(v2bf*)&As[b][arr + i * 16][arc] = pk2(aReg[i].x, aReg[i].y);
      #pragma unroll
      for (int i = 0; i < 2; ++i) {    // adjacent (k,k+1) packed explicitly
        const int k = brk + i * 16;
        *(v2bf*)&Bs[b][brn + 0][k] = pk2(bReg[2 * i].x, bReg[2 * i + 1].x);
        *(v2bf*)&Bs[b][brn + 1][k] = pk2(bReg[2 * i].y, bReg[2 * i + 1].y);
        *(v2bf*)&Bs[b][brn + 2][k] = pk2(bReg[2 * i].z, bReg[2 * i + 1].z);
        *(v2bf*)&Bs[b][brn + 3][k] = pk2(bReg[2 * i].w, bReg[2 * i + 1].w);
      }
    };

    const int Kfull = K & ~(GBK - 1);
    int cur = 0;
    loadRegs(0);
    storeRegs(0);
    __syncthreads();
    for (int kb = 0; kb < Kfull; kb += GBK) {
      const int kn = kb + GBK;
      const bool more = (kn < Kfull);
      if (more) loadRegs(kn);          // overlap with WMMA below
      computeTile(cur);
      if (more) storeRegs(cur ^ 1);    // other buffer: safe before barrier
      __syncthreads();
      cur ^= 1;
    }
    if (Kfull < K) {                   // guarded K tail (one tile)
      for (int i = tid; i < GBM * GBK; i += 256) {
        int r = i >> 5, c = i & 31, gk = Kfull + c;
        As[cur][r][c] = (__bf16)((gk < K) ? A[(long)(m0 + r) * lda + gk] : 0.f);
      }
      for (int i = tid; i < GBK * GBN; i += 256) {
        int c = i >> 7, n = i & 127, gk = Kfull + c;
        Bs[cur][n][c] = (__bf16)((gk < K) ? B[(long)gk * ldb + n0 + n] : 0.f);
      }
      __syncthreads();
      computeTile(cur);
    }
  } else {
    // ---------------- guarded edge path (rare blocks) ----------------
    for (int kb = 0; kb < K; kb += GBK) {
      for (int i = tid; i < GBM * GBK; i += 256) {
        int r = i >> 5, c = i & 31;
        int gm = m0 + r, gk = kb + c;
        As[0][r][c] = (__bf16)((gm < M && gk < K) ? A[(long)gm * lda + gk] : 0.f);
      }
      for (int i = tid; i < GBK * GBN; i += 256) {
        int c = i >> 7, n = i & 127;
        int gk = kb + c, gn = n0 + n;
        Bs[0][n][c] = (__bf16)((gk < K && gn < N) ? B[(long)gk * ldb + gn] : 0.f);
      }
      __syncthreads();
      computeTile(0);
      __syncthreads();
    }
  }

  // store: C layout -> lane n=lane&15, rows g*8+v
  #pragma unroll
  for (int j = 0; j < 2; ++j) {
    const int ncol = n0 + wn + j * 16 + l15;
    if (ncol >= N) continue;
    #pragma unroll
    for (int i = 0; i < 2; ++i) {
      #pragma unroll
      for (int v = 0; v < 8; ++v) {
        const int mrow = m0 + wm + i * 16 + g * 8 + v;
        if (mrow < M) {
          float out = acc[i][j][v];
          if (Cin) out += Cin[(long)mrow * ldc + ncol];
          C[(long)mrow * ldc + ncol] = out;
        }
      }
    }
  }
}

// ------------------------------------------------- q/k/v prep (l2norm etc.)
__global__ __launch_bounds__(64)
void prep_q_kernel(const float* __restrict__ Q, const float* __restrict__ qs,
                   bf16* __restrict__ Qn, int NQ) {
  const int h = blockIdx.x / NQ, i = blockIdx.x % NQ;
  const int d = threadIdx.x;
  const float v = Q[(long)i * DMODEL + h * DH + d];
  float sq = v * v;
  for (int off = 16; off; off >>= 1) sq += __shfl_xor(sq, off, 32);
  __shared__ float ss[2];
  if ((threadIdx.x & 31) == 0) ss[threadIdx.x >> 5] = sq;
  __syncthreads();
  const float inv = 1.f / fmaxf(sqrtf(ss[0] + ss[1]), 1e-12f);
  Qn[((long)h * NQ + i) * DH + d] = (__bf16)(v * inv * qs[d] * QKSCALE);
}

__global__ __launch_bounds__(64)
void prep_k_kernel(const float* __restrict__ KV, const float* __restrict__ nullkv,
                   const float* __restrict__ ks, bf16* __restrict__ Kn,
                   int nk, int nkp) {
  const int h = blockIdx.x / nkp, r = blockIdx.x % nkp;
  const int d = threadIdx.x;
  float v = 0.f;
  if (r == 0)          v = nullkv[(0 * NH + h) * DH + d];            // null k
  else if (r < nk)     v = KV[(long)(r - 1) * (2 * DMODEL) + h * DH + d];
  float sq = v * v;
  for (int off = 16; off; off >>= 1) sq += __shfl_xor(sq, off, 32);
  __shared__ float ss[2];
  if ((threadIdx.x & 31) == 0) ss[threadIdx.x >> 5] = sq;
  __syncthreads();
  const float inv = (r < nk) ? (1.f / fmaxf(sqrtf(ss[0] + ss[1]), 1e-12f)) : 0.f;
  Kn[((long)h * nkp + r) * DH + d] = (__bf16)(v * inv * ks[d]);
}

__global__ __launch_bounds__(64)
void prep_v_kernel(const float* __restrict__ KV, const float* __restrict__ nullkv,
                   bf16* __restrict__ Vt, int nk, int nkp) {
  const int h = blockIdx.x / nkp, r = blockIdx.x % nkp;
  const int d = threadIdx.x;
  float v = 0.f;
  if (r == 0)          v = nullkv[(1 * NH + h) * DH + d];            // null v
  else if (r < nk)     v = KV[(long)(r - 1) * (2 * DMODEL) + DMODEL + h * DH + d];
  Vt[((long)h * DH + d) * nkp + r] = (__bf16)v;                      // [h][d][key]
}

// ------------------------------------------------------ flash attention
// grid: (NQ/128, NH); 8 waves x 16 query rows. Qn [h][q][64] bf16 (l2n*qs*8),
// Kn [h][key][64] bf16 (l2n*ks), Vt [h][d][key] bf16. Out f32 [NQ][1024].
__global__ __launch_bounds__(256)
void attn_flash_kernel(const bf16* __restrict__ Qn, const bf16* __restrict__ Kn,
                       const bf16* __restrict__ Vt, float* __restrict__ Out,
                       int NQ, int nk, int nkp) {
  __shared__ __bf16 Ps[8][16][32] __attribute__((aligned(64)));
  const int tid  = threadIdx.x;
  const int lane = tid & 31;
  const int wave = tid >> 5;
  const int h  = blockIdx.y;
  const int q0 = blockIdx.x * 128 + wave * 16;
  const int g   = lane >> 4;
  const int l15 = lane & 15;

  // Q as two A-fragments (d 0..31, 32..63)
  const bf16* qrow = Qn + ((long)h * NQ + (q0 + l15)) * DH;
  v16bf qa[2];
  #pragma unroll
  for (int c = 0; c < 2; ++c) {
    v8bf lo = *(const v8bf*)(qrow + c * 32 + g * 8);
    v8bf hi = *(const v8bf*)(qrow + c * 32 + 16 + g * 8);
    qa[c] = cat8(lo, hi);
  }

  v8f o[4] = {};
  float mrow[8], lrow[8];
  #pragma unroll
  for (int v = 0; v < 8; ++v) { mrow[v] = -1e30f; lrow[v] = 0.f; }

  const bf16* Kh = Kn + (long)h * nkp * DH;
  const bf16* Vh = Vt + (long)h * DH * nkp;

  for (int j0 = 0; j0 < nk; j0 += 32) {
    // S = Q @ K^T  (two 16x16 tiles: keys j0..+15, j0+16..+31)
    v8f s[2] = {};
    #pragma unroll
    for (int t = 0; t < 2; ++t) {
      const bf16* krow = Kh + (long)(j0 + t * 16 + l15) * DH;
      #pragma unroll
      for (int c = 0; c < 2; ++c) {
        v16bf kb = cat8(*(const v8bf*)(krow + c * 32 + g * 8),
                        *(const v8bf*)(krow + c * 32 + 16 + g * 8));
        s[t] = __builtin_amdgcn_wmma_f32_16x16x32_bf16(
            false, qa[c], false, kb, (short)0, s[t], false, false);
      }
    }

    // online softmax (row stats per lane-group: group g holds rows g*8..g*8+7)
    const int key0 = j0 + l15;
    const int key1 = j0 + 16 + l15;
    #pragma unroll
    for (int v = 0; v < 8; ++v) {
      float a0 = (key0 < nk) ? s[0][v] : -1e30f;
      float a1 = (key1 < nk) ? s[1][v] : -1e30f;
      float bm = fmaxf(a0, a1);
      for (int off = 8; off; off >>= 1) bm = fmaxf(bm, __shfl_xor(bm, off, 32));
      const float mnew  = fmaxf(mrow[v], bm);
      const float scale = __expf(mrow[v] - mnew);
      const float p0 = (key0 < nk) ? __expf(a0 - mnew) : 0.f;
      const float p1 = (key1 < nk) ? __expf(a1 - mnew) : 0.f;
      float ps = p0 + p1;
      for (int off = 8; off; off >>= 1) ps += __shfl_xor(ps, off, 32);
      lrow[v] = lrow[v] * scale + ps;
      mrow[v] = mnew;
      #pragma unroll
      for (int t = 0; t < 4; ++t) o[t][v] *= scale;
      Ps[wave][g * 8 + v][l15]      = (__bf16)p0;   // C-layout -> LDS
      Ps[wave][g * 8 + v][16 + l15] = (__bf16)p1;
    }
    asm volatile("s_wait_dscnt 0" ::: "memory");     // same-wave DS RAW fence

    // reread P in A-layout, accumulate O += P @ V (K-dim = 32 keys)
    v8bf plo = *(const v8bf*)&Ps[wave][l15][g * 8];
    v8bf phi = *(const v8bf*)&Ps[wave][l15][16 + g * 8];
    v16bf pa = cat8(plo, phi);
    #pragma unroll
    for (int t = 0; t < 4; ++t) {
      v16bf vb = cat8(*(const v8bf*)(Vh + (long)(t * 16 + l15) * nkp + j0 + g * 8),
                      *(const v8bf*)(Vh + (long)(t * 16 + l15) * nkp + j0 + 16 + g * 8));
      o[t] = __builtin_amdgcn_wmma_f32_16x16x32_bf16(
          false, pa, false, vb, (short)0, o[t], false, false);
    }
    asm volatile("s_wait_dscnt 0" ::: "memory");     // P reads before next store
  }

  float linv[8];
  #pragma unroll
  for (int v = 0; v < 8; ++v) linv[v] = 1.f / lrow[v];
  #pragma unroll
  for (int t = 0; t < 4; ++t) {
    const int col = h * DH + t * 16 + l15;
    #pragma unroll
    for (int v = 0; v < 8; ++v) {
      const int row = q0 + g * 8 + v;
      Out[(long)row * DMODEL + col] = o[t][v] * linv[v];
    }
  }
}

// ---------------------------------------------------------------- GEGLU
__global__ __launch_bounds__(256)
void geglu_kernel(const float* __restrict__ Hcat, float* __restrict__ Outp, long total) {
  long i = (long)blockIdx.x * 256 + threadIdx.x;
  if (i >= total) return;
  const long row = i / FFIN, col = i % FFIN;
  const float a    = Hcat[row * (long)FF2IN + col];
  const float gate = Hcat[row * (long)FF2IN + FFIN + col];
  const float ge   = 0.5f * a * (1.f + erff(a * 0.70710678118654752f));
  Outp[i] = gate * ge;
}

// ---------------------------------------------------------------- launch
static inline void* wsal(char** p, size_t bytes) {
  void* r = *p;
  *p += (bytes + 255) & ~(size_t)255;
  return r;
}

extern "C" void kernel_launch(void* const* d_in, const int* in_sizes, int n_in,
                              void* d_out, int out_size, void* d_ws, size_t ws_size,
                              hipStream_t stream) {
  (void)in_sizes; (void)n_in; (void)out_size; (void)ws_size;
  const int*   x_ids    = (const int*)  d_in[0];
  const int*   cond_ids = (const int*)  d_in[1];
  const float* tok_emb  = (const float*)d_in[2];
  const float* pos_emb  = (const float*)d_in[3];
  const float* ctok_emb = (const float*)d_in[4];
  const float* cpos_emb = (const float*)d_in[5];
  const float* sa_g     = (const float*)d_in[6];
  const float* sa_wq    = (const float*)d_in[7];
  const float* sa_wkv   = (const float*)d_in[8];
  const float* sa_null  = (const float*)d_in[9];
  const float* sa_qs    = (const float*)d_in[10];
  const float* sa_ks    = (const float*)d_in[11];
  const float* sa_wo    = (const float*)d_in[12];
  const float* ca_g     = (const float*)d_in[13];
  const float* ca_wq    = (const float*)d_in[14];
  const float* ca_wkv   = (const float*)d_in[15];
  const float* ca_null  = (const float*)d_in[16];
  const float* ca_qs    = (const float*)d_in[17];
  const float* ca_ks    = (const float*)d_in[18];
  const float* ca_wo    = (const float*)d_in[19];
  const float* ff_g1    = (const float*)d_in[20];
  const float* ff_w1    = (const float*)d_in[21];
  const float* ff_g2    = (const float*)d_in[22];
  const float* ff_w2    = (const float*)d_in[23];
  const float* final_g  = (const float*)d_in[24];
  const float* w_logits = (const float*)d_in[25];
  float* logits = (float*)d_out;

  char* wp = (char*)d_ws;
  float* h    = (float*)wsal(&wp, (size_t)NTOK  * DMODEL * 4);
  float* ctx  = (float*)wsal(&wp, (size_t)NCOND * DMODEL * 4);
  float* xn   = (float*)wsal(&wp, (size_t)NTOK  * DMODEL * 4);
  float* qb   = (float*)wsal(&wp, (size_t)NTOK  * DMODEL * 4);
  float* kvb  = (float*)wsal(&wp, (size_t)NTOK  * 2 * DMODEL * 4);
  bf16*  qn   = (bf16*) wsal(&wp, (size_t)NH * NTOK   * DH * 2);
  bf16*  kn   = (bf16*) wsal(&wp, (size_t)NH * NKP_SA * DH * 2);
  bf16*  vt   = (bf16*) wsal(&wp, (size_t)NH * DH * NKP_SA * 2);
  float* ao   = (float*)wsal(&wp, (size_t)NTOK * DMODEL * 4);
  float* ffh  = (float*)wsal(&wp, (size_t)NTOK * FF2IN * 4);
  float* ffg  = (float*)wsal(&wp, (size_t)NTOK * FFIN * 4);
  float* ffgn = (float*)wsal(&wp, (size_t)NTOK * FFIN * 4);

  auto gemm = [&](const float* A, const float* B, float* C, const float* Cin,
                  int M, int N, int K, int lda, int ldb, int ldc) {
    dim3 grid((N + GBN - 1) / GBN, (M + GBM - 1) / GBM);
    gemm_bf16_kernel<<<grid, 256, 0, stream>>>(A, B, C, Cin, M, N, K, lda, ldb, ldc);
  };

  // embeddings
  embed_kernel<<<NTOK,  256, 0, stream>>>(x_ids,    tok_emb,  pos_emb,  h);
  embed_kernel<<<NCOND, 256, 0, stream>>>(cond_ids, ctok_emb, cpos_emb, ctx);

  for (int l = 0; l < NLAYER; ++l) {
    const float* swq  = sa_wq  + (long)l * DMODEL * DMODEL;
    const float* swkv = sa_wkv + (long)l * DMODEL * 2 * DMODEL;
    const float* swo  = sa_wo  + (long)l * DMODEL * DMODEL;
    const float* snul = sa_null + (long)l * 2 * NH * DH;
    const float* cwq  = ca_wq  + (long)l * DMODEL * DMODEL;
    const float* cwkv = ca_wkv + (long)l * DMODEL * 2 * DMODEL;
    const float* cwo  = ca_wo  + (long)l * DMODEL * DMODEL;
    const float* cnul = ca_null + (long)l * 2 * NH * DH;
    const float* fw1  = ff_w1 + (long)l * DMODEL * FF2IN;
    const float* fw2  = ff_w2 + (long)l * FFIN * DMODEL;

    // ---- self attention
    ln_kernel<<<NTOK, 256, 0, stream>>>(h, sa_g + l * DMODEL, xn, DMODEL);
    gemm(xn, swq,  qb,  nullptr, NTOK, DMODEL,     DMODEL, DMODEL, DMODEL,     DMODEL);
    gemm(xn, swkv, kvb, nullptr, NTOK, 2 * DMODEL, DMODEL, DMODEL, 2 * DMODEL, 2 * DMODEL);
    prep_q_kernel<<<NH * NTOK,   64, 0, stream>>>(qb,  sa_qs + l * DH, qn, NTOK);
    prep_k_kernel<<<NH * NKP_SA, 64, 0, stream>>>(kvb, snul, sa_ks + l * DH, kn, NK_SA, NKP_SA);
    prep_v_kernel<<<NH * NKP_SA, 64, 0, stream>>>(kvb, snul, vt, NK_SA, NKP_SA);
    {
      dim3 grid(NTOK / 128, NH);
      attn_flash_kernel<<<grid, 256, 0, stream>>>(qn, kn, vt, ao, NTOK, NK_SA, NKP_SA);
    }
    gemm(ao, swo, h, h, NTOK, DMODEL, DMODEL, DMODEL, DMODEL, DMODEL);   // residual

    // ---- cross attention (kv from raw ctx)
    ln_kernel<<<NTOK, 256, 0, stream>>>(h, ca_g + l * DMODEL, xn, DMODEL);
    gemm(xn,  cwq,  qb,  nullptr, NTOK,  DMODEL,     DMODEL, DMODEL, DMODEL,     DMODEL);
    gemm(ctx, cwkv, kvb, nullptr, NCOND, 2 * DMODEL, DMODEL, DMODEL, 2 * DMODEL, 2 * DMODEL);
    prep_q_kernel<<<NH * NTOK,   64, 0, stream>>>(qb,  ca_qs + l * DH, qn, NTOK);
    prep_k_kernel<<<NH * NKP_CA, 64, 0, stream>>>(kvb, cnul, ca_ks + l * DH, kn, NK_CA, NKP_CA);
    prep_v_kernel<<<NH * NKP_CA, 64, 0, stream>>>(kvb, cnul, vt, NK_CA, NKP_CA);
    {
      dim3 grid(NTOK / 128, NH);
      attn_flash_kernel<<<grid, 256, 0, stream>>>(qn, kn, vt, ao, NTOK, NK_CA, NKP_CA);
    }
    gemm(ao, cwo, h, h, NTOK, DMODEL, DMODEL, DMODEL, DMODEL, DMODEL);   // residual

    // ---- GEGLU feed-forward
    ln_kernel<<<NTOK, 256, 0, stream>>>(h, ff_g1 + l * DMODEL, xn, DMODEL);
    gemm(xn, fw1, ffh, nullptr, NTOK, FF2IN, DMODEL, DMODEL, FF2IN, FF2IN);
    {
      long total = (long)NTOK * FFIN;
      long blocks = (total + 255) / 256;
      geglu_kernel<<<(unsigned)blocks, 256, 0, stream>>>(ffh, ffg, total);
    }
    ln_kernel<<<NTOK, 256, 0, stream>>>(ffg, ff_g2 + l * FFIN, ffgn, FFIN);
    gemm(ffgn, fw2, h, h, NTOK, DMODEL, FFIN, FFIN, DMODEL, DMODEL);     // residual
  }

  // final norm + logits
  ln_kernel<<<NTOK, 256, 0, stream>>>(h, final_g, xn, DMODEL);
  gemm(xn, w_logits, logits, nullptr, NTOK, VOC, DMODEL, DMODEL, VOC, VOC);
}
